// DecoderWithAttention_16398185136497
// MI455X (gfx1250) — matrix operand order repaired and checked by reading, not compiled, not run on hardware
//
#include <hip/hip_runtime.h>
#include <hip/hip_bf16.h>

typedef __attribute__((ext_vector_type(16))) _Float16 v16h;
typedef __attribute__((ext_vector_type(8)))  _Float16 v8h;
typedef __attribute__((ext_vector_type(8)))  float    v8f;

#define B_    64
#define P_    196
#define ENC_  2048
#define ATT_  512
#define DEC_  512
#define EMB_  512
#define V_    10000
#define L_    32
#define T_    31
#define XDIM_ (EMB_ + ENC_)   /* 2560 */
#define G4_   (4 * DEC_)      /* 2048 */

// ---------------------------------------------------------------------------
// Register-blocked WMMA helper: one wave computes NT adjacent 16x16 tiles in N,
// sharing a single A fragment per K-chunk.
//   Arow = A + (m0 + lane%16)*lda              (A row-major MxK, f16)
//   Brow = Bt + (n0 + lane%16)*K               (Bt stored N x K, f16)
//   bstride = 16*K                             (halfs between adjacent N tiles)
// Per chunk: 2 contiguous 16B A loads + 2*NT contiguous 16B B loads + NT wmma.
// ---------------------------------------------------------------------------
template <int NT>
__device__ __forceinline__ void wmma_k_loopN(const _Float16* __restrict__ Arow,
                                             const _Float16* __restrict__ Brow,
                                             size_t bstride, int K, int lane,
                                             v8f* acc) {
  const int kh = (lane >> 4) << 3;   // A half-K select: 0 or 8
  const int kb = (lane >> 4) << 3;   // B half-K select: 0 or 8
  for (int k0 = 0; k0 < K; k0 += 32) {
    if (k0 + 32 < K) __builtin_prefetch(Brow + k0 + 32 + kb, 0, 0);
    const v8h* ap = (const v8h*)(Arow + k0 + kh);
    v8h a0 = ap[0];                  // K = k0+kh    .. +7
    v8h a1 = ap[2];                  // K = k0+kh+16 .. +23
    v16h a;
#pragma unroll
    for (int i = 0; i < 8; ++i) { a[i] = a0[i]; a[8 + i] = a1[i]; }
#pragma unroll
    for (int j = 0; j < NT; ++j) {
      const v8h* bp = (const v8h*)(Brow + (size_t)j * bstride + k0 + kb);
      v8h b0 = bp[0], b1 = bp[2];
      v16h b;
#pragma unroll
      for (int i = 0; i < 8; ++i) { b[i] = b0[i]; b[8 + i] = b1[i]; }
      acc[j] = __builtin_amdgcn_wmma_f32_16x16x32_f16(false, a, false, b,
                                                      (short)0, acc[j],
                                                      false, false);
    }
  }
}

// ---------------------------------------------------------------------------
// Setup kernels
// ---------------------------------------------------------------------------
__global__ void prep_kernel(const int* __restrict__ cap_len,
                            const int* __restrict__ caps_in,
                            int* __restrict__ sort_ind, int* __restrict__ dec_len,
                            int* __restrict__ caps_i,
                            float* __restrict__ out_caps,
                            float* __restrict__ out_declen,
                            float* __restrict__ out_sortind) {
  __shared__ int len_s[B_];
  __shared__ int sidx[B_];
  int t = threadIdx.x;
  if (t < B_) len_s[t] = cap_len[t];
  __syncthreads();
  if (t < B_) {
    int my = len_s[t];
    int r = 0;
    for (int j = 0; j < B_; ++j) {
      int lj = len_s[j];
      if (lj > my || (lj == my && j < t)) ++r;   // stable descending
    }
    sidx[r] = t;
  }
  __syncthreads();
  if (t < B_) {
    int src = sidx[t];
    sort_ind[t] = src;
    int dl = len_s[src] - 1;
    dec_len[t] = dl;
    out_declen[t] = (float)dl;
    out_sortind[t] = (float)src;
    for (int l = 0; l < L_; ++l) {
      int cv = caps_in[src * L_ + l];
      caps_i[t * L_ + l] = cv;
      out_caps[t * L_ + l] = (float)cv;
    }
  }
}

__global__ void cvt_f16_kernel(_Float16* __restrict__ dst,
                               const float* __restrict__ src, int n) {
  int i = blockIdx.x * blockDim.x + threadIdx.x;
  if (i < n) dst[i] = (_Float16)src[i];
}

// src: R x C row-major  ->  dst: C x R row-major (transpose + f16 convert)
__global__ void cvt_t_f16_kernel(_Float16* __restrict__ dst,
                                 const float* __restrict__ src, int R, int C) {
  int i = blockIdx.x * blockDim.x + threadIdx.x;
  if (i < R * C) {
    int c = i / R, r = i % R;
    dst[i] = (_Float16)src[(size_t)r * C + c];
  }
}

// att1 = enc_sorted @ W_enc_att + b  (M=12544, K=2048, N=512)
// A is f32 (converted in-register once per chunk, shared by 4 N tiles).
__global__ void att1_gemm_kernel(const float* __restrict__ enc,
                                 const int* __restrict__ sort_ind,
                                 const _Float16* __restrict__ WencT,
                                 const float* __restrict__ bias,
                                 float* __restrict__ att1) {
  int lane = threadIdx.x;
  int n0 = blockIdx.x * 64;   // 4 tiles of 16
  int m0 = blockIdx.y * 16;   // 0..12543
  int mrow = m0 + (lane & 15);
  int b = mrow / P_, pp = mrow % P_;
  const float* arow = enc + ((size_t)sort_ind[b] * P_ + pp) * ENC_;
  const _Float16* Brow = WencT + (size_t)(n0 + (lane & 15)) * ENC_;
  const int kh = (lane >> 4) << 3;
  const int kb = (lane >> 4) << 3;
  v8f acc[4] = {};
  for (int k0 = 0; k0 < ENC_; k0 += 32) {
    if (k0 + 32 < ENC_) __builtin_prefetch(arow + k0 + 32 + kh, 0, 0);
    const float4* ap = (const float4*)(arow + k0 + kh);
    float4 f0 = ap[0], f1 = ap[1];   // K = kh..kh+7
    float4 f2 = ap[4], f3 = ap[5];   // K = kh+16..kh+23
    v16h a;
    a[0] = (_Float16)f0.x; a[1] = (_Float16)f0.y; a[2] = (_Float16)f0.z; a[3] = (_Float16)f0.w;
    a[4] = (_Float16)f1.x; a[5] = (_Float16)f1.y; a[6] = (_Float16)f1.z; a[7] = (_Float16)f1.w;
    a[8]  = (_Float16)f2.x; a[9]  = (_Float16)f2.y; a[10] = (_Float16)f2.z; a[11] = (_Float16)f2.w;
    a[12] = (_Float16)f3.x; a[13] = (_Float16)f3.y; a[14] = (_Float16)f3.z; a[15] = (_Float16)f3.w;
#pragma unroll
    for (int j = 0; j < 4; ++j) {
      const v8h* bp = (const v8h*)(Brow + (size_t)j * 16 * ENC_ + k0 + kb);
      v8h b0 = bp[0], b1 = bp[2];
      v16h bf;
#pragma unroll
      for (int i = 0; i < 8; ++i) { bf[i] = b0[i]; bf[8 + i] = b1[i]; }
      acc[j] = __builtin_amdgcn_wmma_f32_16x16x32_f16(false, a, false, bf,
                                                      (short)0, acc[j],
                                                      false, false);
    }
  }
  int moff = (lane >> 4) << 3;
#pragma unroll
  for (int j = 0; j < 4; ++j) {
    int n = n0 + 16 * j + (lane & 15);
    float bv = bias[n];
#pragma unroll
    for (int r = 0; r < 8; ++r)
      att1[(size_t)(m0 + moff + r) * ATT_ + n] = acc[j][r] + bv;
  }
}

__global__ void mean_enc_kernel(const float* __restrict__ enc,
                                const int* __restrict__ sort_ind,
                                float* __restrict__ mean_enc) {
  int e = blockIdx.x * blockDim.x + threadIdx.x;   // < 2048
  int b = blockIdx.y;
  const float* base = enc + (size_t)sort_ind[b] * P_ * ENC_ + e;
  float s = 0.f;
  for (int p = 0; p < P_; ++p) s += base[(size_t)p * ENC_];
  mean_enc[b * ENC_ + e] = s * (1.0f / (float)P_);
}

__global__ void init_hc_kernel(const float* __restrict__ mean_enc,
                               const float* __restrict__ Wih_, const float* __restrict__ bih_,
                               const float* __restrict__ Wic_, const float* __restrict__ bic_,
                               float* __restrict__ h, float* __restrict__ c,
                               _Float16* __restrict__ h16) {
  int d = blockIdx.x * blockDim.x + threadIdx.x;   // < 512
  int b = blockIdx.y;
  const float* me = mean_enc + b * ENC_;
  float sh = bih_[d], sc = bic_[d];
  for (int e = 0; e < ENC_; ++e) {
    float v = me[e];
    sh += v * Wih_[(size_t)e * DEC_ + d];
    sc += v * Wic_[(size_t)e * DEC_ + d];
  }
  h[b * DEC_ + d] = sh;
  c[b * DEC_ + d] = sc;
  h16[b * DEC_ + d] = (_Float16)sh;
}

// ---------------------------------------------------------------------------
// Per-step kernels
// ---------------------------------------------------------------------------
__global__ void gemm_att2_kernel(const _Float16* __restrict__ h16,
                                 const _Float16* __restrict__ WdecT,
                                 const float* __restrict__ bias,
                                 float* __restrict__ C) {
  int lane = threadIdx.x;
  int n0 = blockIdx.x * 64, m0 = blockIdx.y * 16;
  v8f acc[4] = {};
  wmma_k_loopN<4>(h16 + (size_t)(m0 + (lane & 15)) * DEC_,
                  WdecT + (size_t)(n0 + (lane & 15)) * DEC_,
                  (size_t)16 * DEC_, DEC_, lane, acc);
  int moff = (lane >> 4) << 3;
#pragma unroll
  for (int j = 0; j < 4; ++j) {
    int n = n0 + 16 * j + (lane & 15);
    float bv = bias[n];
#pragma unroll
    for (int r = 0; r < 8; ++r)
      C[(size_t)(m0 + moff + r) * ATT_ + n] = acc[j][r] + bv;
  }
}

__global__ void gemm_fbeta_kernel(const _Float16* __restrict__ h16,
                                  const _Float16* __restrict__ WfbT,
                                  const float* __restrict__ bias,
                                  float* __restrict__ gate) {
  int lane = threadIdx.x;
  int n0 = blockIdx.x * 64, m0 = blockIdx.y * 16;
  v8f acc[4] = {};
  wmma_k_loopN<4>(h16 + (size_t)(m0 + (lane & 15)) * DEC_,
                  WfbT + (size_t)(n0 + (lane & 15)) * DEC_,
                  (size_t)16 * DEC_, DEC_, lane, acc);
  int moff = (lane >> 4) << 3;
#pragma unroll
  for (int j = 0; j < 4; ++j) {
    int n = n0 + 16 * j + (lane & 15);
    float bv = bias[n];
#pragma unroll
    for (int r = 0; r < 8; ++r) {
      float v = acc[j][r] + bv;
      gate[(size_t)(m0 + moff + r) * ENC_ + n] = 1.0f / (1.0f + __expf(-v));
    }
  }
}

__global__ void score_softmax_kernel(const float* __restrict__ att1,
                                     const float* __restrict__ att2,
                                     const float* __restrict__ w_full,
                                     const float* __restrict__ b_full,
                                     const int* __restrict__ dec_len, int t,
                                     float* __restrict__ alpha,
                                     float* __restrict__ alphas_out) {
  int b = blockIdx.x;
  int tid = threadIdx.x;
  __shared__ float a2s[ATT_];
  __shared__ float sc[P_];
  __shared__ float red[256];
  // Async-stage att2 row (2KB) into LDS: 128 lanes x b128, ASYNCcnt-tracked.
  if (tid < 128) {
    unsigned long long gp =
        (unsigned long long)(uintptr_t)(att2 + (size_t)b * ATT_ + tid * 4);
    unsigned lo = (unsigned)(uintptr_t)(&a2s[tid * 4]);
    asm volatile("global_load_async_to_lds_b128 %0, %1, off"
                 :: "v"(lo), "v"(gp) : "memory");
  }
  asm volatile("s_wait_asynccnt 0x0" ::: "memory");
  __syncthreads();
  const float* a1 = att1 + (size_t)b * P_ * ATT_;
  for (int p = tid; p < P_; p += 256) {
    const float* row = a1 + (size_t)p * ATT_;
    float s = 0.f;
    for (int k = 0; k < ATT_; ++k) {
      float v = row[k] + a2s[k];
      v = v > 0.f ? v : 0.f;
      s += v * w_full[k];
    }
    sc[p] = s + b_full[0];
  }
  __syncthreads();
  float m = -1e30f;
  for (int p = tid; p < P_; p += 256) m = fmaxf(m, sc[p]);
  red[tid] = m;
  __syncthreads();
  for (int s = 128; s > 0; s >>= 1) {
    if (tid < s) red[tid] = fmaxf(red[tid], red[tid + s]);
    __syncthreads();
  }
  float mx = red[0];
  __syncthreads();
  float e = 0.f;
  for (int p = tid; p < P_; p += 256) {
    float v = __expf(sc[p] - mx);
    sc[p] = v;
    e += v;
  }
  red[tid] = e;
  __syncthreads();
  for (int s = 128; s > 0; s >>= 1) {
    if (tid < s) red[tid] += red[tid + s];
    __syncthreads();
  }
  float inv = 1.0f / red[0];
  float mflt = (t < dec_len[b]) ? 1.0f : 0.0f;
  for (int p = tid; p < P_; p += 256) {
    float al = sc[p] * inv;
    alpha[b * P_ + p] = al;
    alphas_out[((size_t)b * T_ + t) * P_ + p] = al * mflt;
  }
}

__global__ void awe_kernel(const float* __restrict__ enc,
                           const int* __restrict__ sort_ind,
                           const float* __restrict__ alpha,
                           float* __restrict__ awe) {
  int e = blockIdx.x * blockDim.x + threadIdx.x;   // < 2048
  int b = blockIdx.y;
  const float* base = enc + (size_t)sort_ind[b] * P_ * ENC_ + e;
  const float* al = alpha + b * P_;
  float s = 0.f;
  for (int p = 0; p < P_; ++p) s += al[p] * base[(size_t)p * ENC_];
  awe[b * ENC_ + e] = s;
}

__global__ void build_x_kernel(const float* __restrict__ embedding,
                               const int* __restrict__ caps_i, int t,
                               const float* __restrict__ gate,
                               const float* __restrict__ awe,
                               _Float16* __restrict__ x16) {
  int j = blockIdx.x * blockDim.x + threadIdx.x;   // < 2560
  int b = blockIdx.y;
  float v;
  if (j < EMB_) {
    v = embedding[(size_t)caps_i[b * L_ + t] * EMB_ + j];
  } else {
    int e = j - EMB_;
    v = gate[b * ENC_ + e] * awe[b * ENC_ + e];
  }
  x16[b * XDIM_ + j] = (_Float16)v;
}

// gates = x @ W_ih^T + h @ W_hh^T + b_ih + b_hh  (M=64, N=2048, K=2560 and 512)
// W_ih / W_hh are already stored N x K by the reference -> straight f16 copy.
__global__ void gemm_gates_kernel(const _Float16* __restrict__ x16,
                                  const _Float16* __restrict__ Wih16,
                                  const _Float16* __restrict__ h16,
                                  const _Float16* __restrict__ Whh16,
                                  const float* __restrict__ bih,
                                  const float* __restrict__ bhh,
                                  float* __restrict__ gates) {
  int lane = threadIdx.x;
  int n0 = blockIdx.x * 64, m0 = blockIdx.y * 16;
  v8f acc[4] = {};
  wmma_k_loopN<4>(x16 + (size_t)(m0 + (lane & 15)) * XDIM_,
                  Wih16 + (size_t)(n0 + (lane & 15)) * XDIM_,
                  (size_t)16 * XDIM_, XDIM_, lane, acc);
  wmma_k_loopN<4>(h16 + (size_t)(m0 + (lane & 15)) * DEC_,
                  Whh16 + (size_t)(n0 + (lane & 15)) * DEC_,
                  (size_t)16 * DEC_, DEC_, lane, acc);
  int moff = (lane >> 4) << 3;
#pragma unroll
  for (int j = 0; j < 4; ++j) {
    int n = n0 + 16 * j + (lane & 15);
    float bv = bih[n] + bhh[n];
#pragma unroll
    for (int r = 0; r < 8; ++r)
      gates[(size_t)(m0 + moff + r) * G4_ + n] = acc[j][r] + bv;
  }
}

__global__ void lstm_update_kernel(const float* __restrict__ gates,
                                   const int* __restrict__ dec_len, int t,
                                   float* __restrict__ h, float* __restrict__ c,
                                   _Float16* __restrict__ h16) {
  int d = blockIdx.x * blockDim.x + threadIdx.x;   // < 512
  int b = blockIdx.y;
  const float* g = gates + (size_t)b * G4_;
  float gi = g[d], gf = g[DEC_ + d], gg = g[2 * DEC_ + d], go = g[3 * DEC_ + d];
  float si = 1.0f / (1.0f + __expf(-gi));
  float sf = 1.0f / (1.0f + __expf(-gf));
  float so = 1.0f / (1.0f + __expf(-go));
  float tg = tanhf(gg);
  float cold = c[b * DEC_ + d], hold = h[b * DEC_ + d];
  float cn = sf * cold + si * tg;
  float hn = so * tanhf(cn);
  bool mk = t < dec_len[b];
  float h2 = mk ? hn : hold;
  float c2 = mk ? cn : cold;
  h[b * DEC_ + d] = h2;
  c[b * DEC_ + d] = c2;
  h16[b * DEC_ + d] = (_Float16)h2;   // masked rows produce 0 in preds anyway
}

// preds[b][t][:] = mask * (h_new @ W_fc + b_fc)  (M=64, N=10000 = 125*5 tiles, K=512)
__global__ void gemm_preds_kernel(const _Float16* __restrict__ h16,
                                  const _Float16* __restrict__ WfcT,
                                  const float* __restrict__ bfc,
                                  const int* __restrict__ dec_len, int t,
                                  float* __restrict__ preds_out) {
  int lane = threadIdx.x;
  int n0 = blockIdx.x * 80;    // 5 tiles of 16
  int m0 = blockIdx.y * 16;
  v8f acc[5] = {};
  wmma_k_loopN<5>(h16 + (size_t)(m0 + (lane & 15)) * DEC_,
                  WfcT + (size_t)(n0 + (lane & 15)) * DEC_,
                  (size_t)16 * DEC_, DEC_, lane, acc);
  int moff = (lane >> 4) << 3;
#pragma unroll
  for (int j = 0; j < 5; ++j) {
    int n = n0 + 16 * j + (lane & 15);
    float bv = bfc[n];
#pragma unroll
    for (int r = 0; r < 8; ++r) {
      int b = m0 + moff + r;
      float mflt = (t < dec_len[b]) ? 1.0f : 0.0f;
      preds_out[((size_t)b * T_ + t) * V_ + n] = (acc[j][r] + bv) * mflt;
    }
  }
}

// ---------------------------------------------------------------------------
// Host launcher
// ---------------------------------------------------------------------------
extern "C" void kernel_launch(void* const* d_in, const int* in_sizes, int n_in,
                              void* d_out, int out_size, void* d_ws, size_t ws_size,
                              hipStream_t stream) {
  (void)in_sizes; (void)n_in; (void)out_size; (void)ws_size;
  const float* encoder_out      = (const float*)d_in[0];
  const int*   encoded_captions = (const int*)d_in[1];
  const int*   caption_lengths  = (const int*)d_in[2];
  const float* W_enc_att = (const float*)d_in[3];
  const float* b_enc_att = (const float*)d_in[4];
  const float* W_dec_att = (const float*)d_in[5];
  const float* b_dec_att = (const float*)d_in[6];
  const float* w_full    = (const float*)d_in[7];
  const float* b_full    = (const float*)d_in[8];
  const float* embedding = (const float*)d_in[9];
  const float* W_ih      = (const float*)d_in[10];
  const float* b_ih      = (const float*)d_in[11];
  const float* W_hh      = (const float*)d_in[12];
  const float* b_hh      = (const float*)d_in[13];
  const float* W_init_h  = (const float*)d_in[14];
  const float* b_init_h  = (const float*)d_in[15];
  const float* W_init_c  = (const float*)d_in[16];
  const float* b_init_c  = (const float*)d_in[17];
  const float* W_fbeta   = (const float*)d_in[18];
  const float* b_fbeta   = (const float*)d_in[19];
  const float* W_fc      = (const float*)d_in[20];
  const float* b_fc      = (const float*)d_in[21];

  // output layout: predictions | caps | dec_len | alphas | sort_ind
  float* out          = (float*)d_out;
  float* out_preds    = out;
  float* out_caps     = out_preds + (size_t)B_ * T_ * V_;
  float* out_declen   = out_caps + B_ * L_;
  float* out_alphas   = out_declen + B_;
  float* out_sortind  = out_alphas + (size_t)B_ * T_ * P_;

  // workspace carve-up (256B aligned blocks -> all f16 panels 16B aligned)
  char* wp = (char*)d_ws;
  auto alloc = [&](size_t bytes) -> void* {
    void* r = (void*)wp;
    wp += (bytes + 255) & ~((size_t)255);
    return r;
  };
  int*      sort_ind = (int*)alloc(B_ * 4);
  int*      dec_len  = (int*)alloc(B_ * 4);
  int*      caps_i   = (int*)alloc(B_ * L_ * 4);
  float*    att1     = (float*)alloc((size_t)B_ * P_ * ATT_ * 4);
  float*    att2     = (float*)alloc((size_t)B_ * ATT_ * 4);
  float*    alpha    = (float*)alloc((size_t)B_ * P_ * 4);
  float*    awe      = (float*)alloc((size_t)B_ * ENC_ * 4);
  float*    gatev    = (float*)alloc((size_t)B_ * ENC_ * 4);
  float*    gates    = (float*)alloc((size_t)B_ * G4_ * 4);
  float*    hbuf     = (float*)alloc((size_t)B_ * DEC_ * 4);
  float*    cbuf     = (float*)alloc((size_t)B_ * DEC_ * 4);
  float*    meanenc  = (float*)alloc((size_t)B_ * ENC_ * 4);
  _Float16* h16      = (_Float16*)alloc((size_t)B_ * DEC_ * 2);
  _Float16* x16      = (_Float16*)alloc((size_t)B_ * XDIM_ * 2);
  _Float16* WencT16  = (_Float16*)alloc((size_t)ATT_ * ENC_ * 2);   // N x K
  _Float16* WdecT16  = (_Float16*)alloc((size_t)ATT_ * DEC_ * 2);   // N x K
  _Float16* WfbT16   = (_Float16*)alloc((size_t)ENC_ * DEC_ * 2);   // N x K
  _Float16* Wih16    = (_Float16*)alloc((size_t)G4_ * XDIM_ * 2);   // already N x K
  _Float16* Whh16    = (_Float16*)alloc((size_t)G4_ * DEC_ * 2);    // already N x K
  _Float16* WfcT16   = (_Float16*)alloc((size_t)V_ * DEC_ * 2);     // N x K

  // ---- setup ----
  prep_kernel<<<1, 64, 0, stream>>>(caption_lengths, encoded_captions,
                                    sort_ind, dec_len, caps_i,
                                    out_caps, out_declen, out_sortind);

  auto cvt = [&](_Float16* dst, const float* src, int n) {
    cvt_f16_kernel<<<(n + 255) / 256, 256, 0, stream>>>(dst, src, n);
  };
  // K x N weights -> transpose to N x K
  cvt_t_f16_kernel<<<(ENC_ * ATT_ + 255) / 256, 256, 0, stream>>>(WencT16, W_enc_att, ENC_, ATT_);
  cvt_t_f16_kernel<<<(DEC_ * ATT_ + 255) / 256, 256, 0, stream>>>(WdecT16, W_dec_att, DEC_, ATT_);
  cvt_t_f16_kernel<<<(DEC_ * ENC_ + 255) / 256, 256, 0, stream>>>(WfbT16, W_fbeta, DEC_, ENC_);
  cvt_t_f16_kernel<<<(DEC_ * V_ + 255) / 256, 256, 0, stream>>>(WfcT16, W_fc, DEC_, V_);
  // already N x K -> plain convert
  cvt(Wih16, W_ih, G4_ * XDIM_);
  cvt(Whh16, W_hh, G4_ * DEC_);

  // att1 GEMM: M=12544 (784 tiles), N=512 (8 blocks x 4 tiles)
  att1_gemm_kernel<<<dim3(ATT_ / 64, (B_ * P_) / 16), 32, 0, stream>>>(
      encoder_out, sort_ind, WencT16, b_enc_att, att1);

  mean_enc_kernel<<<dim3(ENC_ / 256, B_), 256, 0, stream>>>(encoder_out, sort_ind, meanenc);
  init_hc_kernel<<<dim3(DEC_ / 256, B_), 256, 0, stream>>>(
      meanenc, W_init_h, b_init_h, W_init_c, b_init_c, hbuf, cbuf, h16);

  // ---- decode loop ----
  for (int t = 0; t < T_; ++t) {
    gemm_att2_kernel<<<dim3(ATT_ / 64, B_ / 16), 32, 0, stream>>>(h16, WdecT16, b_dec_att, att2);
    gemm_fbeta_kernel<<<dim3(ENC_ / 64, B_ / 16), 32, 0, stream>>>(h16, WfbT16, b_fbeta, gatev);
    score_softmax_kernel<<<B_, 256, 0, stream>>>(att1, att2, w_full, b_full,
                                                 dec_len, t, alpha, out_alphas);
    awe_kernel<<<dim3(ENC_ / 256, B_), 256, 0, stream>>>(encoder_out, sort_ind, alpha, awe);
    build_x_kernel<<<dim3(XDIM_ / 256, B_), 256, 0, stream>>>(embedding, caps_i, t,
                                                              gatev, awe, x16);
    gemm_gates_kernel<<<dim3(G4_ / 64, B_ / 16), 32, 0, stream>>>(
        x16, Wih16, h16, Whh16, b_ih, b_hh, gates);
    lstm_update_kernel<<<dim3(DEC_ / 256, B_), 256, 0, stream>>>(gates, dec_len, t,
                                                                 hbuf, cbuf, h16);
    gemm_preds_kernel<<<dim3(V_ / 80, B_ / 16), 32, 0, stream>>>(
        h16, WfcT16, b_fc, dec_len, t, out_preds);
  }
}
